// MiniGRUParallelCell_43379169689745
// MI455X (gfx1250) — matrix operand order repaired and checked by reading, not compiled, not run on hardware
//
#include <hip/hip_runtime.h>

// Problem constants (from reference): B=8, T=4096, DIN=DOUT=512
#define NB   8
#define NT   4096
#define ND   512
#define TP1  4097          // T+1 output time steps
#define LCH  64            // time-scan chunk length
#define NCH  65            // ceil(4097/64)
#define NKT  16            // ND/32 k-tiles
#define NNT  32            // ND/16 n-tiles

typedef __attribute__((ext_vector_type(16))) __bf16 v16bf;
typedef __attribute__((ext_vector_type(8)))  float  v8f;

__device__ __forceinline__ float softplusf(float x) {
  // stable log(1+exp(x))
  if (x > 20.0f)  return x;
  if (x < -20.0f) return expf(x);
  return log1pf(expf(x));
}
__device__ __forceinline__ float log_gf(float x) {
  // log(relu(x)+0.5) for x>=0 else -softplus(-x)
  return (x >= 0.0f) ? logf(x + 0.5f) : -softplusf(-x);
}

// ---------------------------------------------------------------------------
// Weight prep: split f32 W[o][d] into bf16 hi/lo, stored in WMMA B-fragment
// layout so each lane's 16 values are one contiguous 32B chunk:
//   frag (nt, kt): lane l covers N = nt*16 + l%16, K = kt*32 + (l/16)*16 + j
//   packed index   = (((nt*NKT) + kt)*32 + l)*16 + j
// Done ONCE; stage 1 then streams pure bf16 fragments from L2.
// ---------------------------------------------------------------------------
__global__ void prep_weights(const float* __restrict__ W,
                             __bf16* __restrict__ hi, __bf16* __restrict__ lo) {
  int idx  = blockIdx.x * blockDim.x + threadIdx.x;   // 0 .. ND*ND-1
  int j    = idx & 15;
  int lane = (idx >> 4) & 31;
  int kt   = (idx >> 9) & (NKT - 1);
  int nt   = idx >> 13;
  int n = nt * 16 + (lane & 15);
  int k = kt * 32 + (lane >> 4) * 16 + j;
  float v  = W[n * ND + k];
  __bf16 h = (__bf16)v;
  hi[idx] = h;
  lo[idx] = (__bf16)(v - (float)h);
}

// ---------------------------------------------------------------------------
// Stage 0: t=0 row: log_values[b,0,o] = log_g(h0[b,o]); a_star[b,0,o] = 0
// ---------------------------------------------------------------------------
__global__ void stage0_init(const float* __restrict__ h0,
                            float* __restrict__ lv, float* __restrict__ as_) {
  int i = blockIdx.x * blockDim.x + threadIdx.x;   // 0..NB*ND-1
  int b = i >> 9, o = i & (ND - 1);
  size_t idx = (size_t)b * TP1 * ND + o;
  lv[idx]  = log_gf(h0[i]);
  as_[idx] = 0.0f;
}

// ---------------------------------------------------------------------------
// Stage 1: fused dual GEMM (split-bf16 WMMA, prepacked weights) + elementwise
// + feature cumsum. One workgroup per (b, 16-row time tile). 512 thr = 16 waves.
// Wave w owns output columns [32w, 32w+32) as two 16x16 WMMA tiles.
// x tile is staged in LDS in A-FRAGMENT order: chunk (kt, lane) holds the 16
// contiguous bf16 a-values for that lane/k-step -> one 32B LDS read each.
//   A layout (16-bit, 16x32): lane: M = lane%16;
//     K = kt*32 + (lane/16)*16 + j (matches B fragment k order)
// ---------------------------------------------------------------------------
__global__ __launch_bounds__(512)
void stage1_gemm(const float* __restrict__ x,
                 const v16bf* __restrict__ Wzh, const v16bf* __restrict__ Wzl,
                 const v16bf* __restrict__ Whh, const v16bf* __restrict__ Whl,
                 const float* __restrict__ bz,  const float* __restrict__ bh,
                 float* __restrict__ lv, float* __restrict__ as_) {
  __shared__ __align__(32) __bf16 sa_hi[16 * ND];  // packed A fragments, hi
  __shared__ __align__(32) __bf16 sa_lo[16 * ND];  // packed A fragments, lo
  __shared__ float s_lc[16][ND];                   // log_coeffs for cumsum

  const int blk  = blockIdx.x;       // 0 .. NB*(NT/16)-1
  const int b    = blk >> 8;         // / (NT/16 = 256)
  const int t0   = (blk & 255) << 4; // time tile start
  const int tid  = threadIdx.x;
  const int wave = tid >> 5;
  const int lane = tid & 31;
  const int g    = lane >> 4;        // half-wave group (0/1)
  const int r    = lane & 15;

  // Cooperative load of 16x512 x tile -> split bf16 hi/lo, A-fragment order.
  const float* xblk = x + ((size_t)b * NT + t0) * ND;
  for (int i = tid; i < 16 * ND; i += 512) {
    int kt_   = i >> 9;          // k-tile
    int lane_ = (i >> 4) & 31;   // lane within fragment
    int j     = i & 15;          // element within lane chunk
    int row   = lane_ & 15;
    int gg    = lane_ >> 4;
    int k     = kt_ * 32 + gg * 16 + j;
    float v  = xblk[row * ND + k];
    __bf16 h = (__bf16)v;
    sa_hi[i] = h;
    sa_lo[i] = (__bf16)(v - (float)h);
  }
  __syncthreads();

  const v16bf* pah = (const v16bf*)sa_hi;
  const v16bf* pal = (const v16bf*)sa_lo;

  const int n0 = wave << 5;           // first of 32 output columns for this wave
  v8f accz[2] = {{}, {}};             // Wz-path accumulators (two N-tiles)
  v8f acch[2] = {{}, {}};             // Wh-path accumulators
  const int nt0 = wave * 2;           // first n-tile index for this wave

  for (int kt = 0; kt < NKT; ++kt) {
    // A fragments: single contiguous 32B chunk per lane -> 2x ds_load_b128
    v16bf a_hi = pah[kt * 32 + lane];
    v16bf a_lo = pal[kt * 32 + lane];
#pragma unroll
    for (int tile = 0; tile < 2; ++tile) {
      const int f = ((nt0 + tile) * NKT + kt) * 32 + lane;
      // Prefetch next k-step's fragments into cache (global_prefetch_b8)
      if (kt + 1 < NKT) {
        __builtin_prefetch(&Wzh[f + 32], 0, 3);
        __builtin_prefetch(&Whh[f + 32], 0, 3);
      }
      v16bf bzh = Wzh[f], bzl = Wzl[f];
      v16bf bhh = Whh[f], bhl = Whl[f];
      // split-precision accumulation: hi*hi + hi*lo + lo*hi
      accz[tile] = __builtin_amdgcn_wmma_f32_16x16x32_bf16(false, a_hi, false, bzh, (short)0, accz[tile], false, false);
      accz[tile] = __builtin_amdgcn_wmma_f32_16x16x32_bf16(false, a_hi, false, bzl, (short)0, accz[tile], false, false);
      accz[tile] = __builtin_amdgcn_wmma_f32_16x16x32_bf16(false, a_lo, false, bzh, (short)0, accz[tile], false, false);
      acch[tile] = __builtin_amdgcn_wmma_f32_16x16x32_bf16(false, a_hi, false, bhh, (short)0, acch[tile], false, false);
      acch[tile] = __builtin_amdgcn_wmma_f32_16x16x32_bf16(false, a_hi, false, bhl, (short)0, acch[tile], false, false);
      acch[tile] = __builtin_amdgcn_wmma_f32_16x16x32_bf16(false, a_lo, false, bhh, (short)0, acch[tile], false, false);
    }
  }

  // Epilogue: bias, softplus/log_g, write log_values, stage log_coeffs
  // C/D layout: lane: N = lane%16; element i -> M = g*8 + i
#pragma unroll
  for (int tile = 0; tile < 2; ++tile) {
    const int ncol  = n0 + tile * 16 + r;
    const float bzv = bz[ncol];
    const float bhv = bh[ncol];
#pragma unroll
    for (int i = 0; i < 8; ++i) {
      int   m  = g * 8 + i;                     // time row within tile
      float kk = accz[tile][i] + bzv;           // z-gate pre-activation
      float hh = acch[tile][i] + bhv;           // candidate pre-activation
      float lvv = -softplusf(-kk) + log_gf(hh); // log_z + log_tilde_h
      s_lc[m][ncol] = -softplusf(kk);           // log_coeffs
      lv[((size_t)b * TP1 + (t0 + m + 1)) * ND + ncol] = lvv;
    }
  }
  __syncthreads();

  // Feature-dim inclusive cumsum of log_coeffs (reference cumsum axis=-1).
  // Wave w scans row w (512 values) via wave32 Hillis-Steele + carry.
  {
    const int row = wave;
    float carry = 0.0f;
    for (int c = 0; c < ND; c += 32) {
      float v = s_lc[row][c + lane];
#pragma unroll
      for (int off = 1; off < 32; off <<= 1) {
        float nb = __shfl_up(v, off, 32);
        if (lane >= off) v += nb;
      }
      v += carry;
      as_[((size_t)b * TP1 + (t0 + row + 1)) * ND + c + lane] = v;
      carry = __shfl(v, 31, 32);
    }
  }
}

// ---------------------------------------------------------------------------
// Time-axis cumlogsumexp of (lv - as), 3-phase chunked scan.
// Thread = one (b, o); block = all 512 o for one (b, chunk). Coalesced.
// ---------------------------------------------------------------------------
__global__ __launch_bounds__(512)
void scan_partial(const float* __restrict__ lv, const float* __restrict__ as_,
                  float* __restrict__ Mc, float* __restrict__ Sc) {
  int b = blockIdx.x / NCH, c = blockIdx.x % NCH, o = threadIdx.x;
  int tbeg = c * LCH, tend = tbeg + LCH; if (tend > TP1) tend = TP1;
  float M = -1e30f, S = 0.0f;
  for (int t = tbeg; t < tend; ++t) {
    size_t idx = ((size_t)b * TP1 + t) * ND + o;
    float u  = lv[idx] - as_[idx];
    float nm = fmaxf(M, u);
    S = S * expf(M - nm) + expf(u - nm);
    M = nm;
  }
  size_t ci = ((size_t)b * NCH + c) * ND + o;
  Mc[ci] = M; Sc[ci] = S;
}

__global__ void scan_spine(const float* __restrict__ Mc, const float* __restrict__ Sc,
                           float* __restrict__ Mp, float* __restrict__ Sp) {
  int i = blockIdx.x * blockDim.x + threadIdx.x;   // 0..NB*ND-1
  int b = i >> 9, o = i & (ND - 1);
  float M = -1e30f, S = 0.0f;                       // "empty" logsumexp state
  for (int c = 0; c < NCH; ++c) {
    size_t ci = ((size_t)b * NCH + c) * ND + o;
    Mp[ci] = M; Sp[ci] = S;                         // exclusive prefix
    float mc = Mc[ci], sc = Sc[ci];
    float nm = fmaxf(M, mc);
    S = S * expf(M - nm) + sc * expf(mc - nm);
    M = nm;
  }
}

__global__ __launch_bounds__(512)
void scan_final(const float* __restrict__ lv, const float* __restrict__ as_,
                const float* __restrict__ Mp, const float* __restrict__ Sp,
                float* __restrict__ out) {
  int b = blockIdx.x / NCH, c = blockIdx.x % NCH, o = threadIdx.x;
  size_t ci = ((size_t)b * NCH + c) * ND + o;
  float M = Mp[ci], S = Sp[ci];
  int tbeg = c * LCH, tend = tbeg + LCH; if (tend > TP1) tend = TP1;
  for (int t = tbeg; t < tend; ++t) {
    size_t idx = ((size_t)b * TP1 + t) * ND + o;
    float a  = as_[idx];
    float u  = lv[idx] - a;
    float nm = fmaxf(M, u);
    S = S * expf(M - nm) + expf(u - nm);
    M = nm;
    out[idx] = expf(a + M + logf(S));   // exp(log_h)
  }
}

// ---------------------------------------------------------------------------
extern "C" void kernel_launch(void* const* d_in, const int* in_sizes, int n_in,
                              void* d_out, int out_size, void* d_ws, size_t ws_size,
                              hipStream_t stream) {
  const float* x  = (const float*)d_in[0];
  const float* h0 = (const float*)d_in[1];
  const float* Wz = (const float*)d_in[2];
  const float* bz = (const float*)d_in[3];
  const float* Wh = (const float*)d_in[4];
  const float* bh = (const float*)d_in[5];
  float* out = (float*)d_out;

  // Workspace layout: lv[NB*TP1*ND] f32, as[NB*TP1*ND] f32,
  //                   Mc,Sc,Mp,Sp each [NB*NCH*ND] f32,
  //                   then 4x packed bf16 weights [ND*ND] each (32B aligned).
  const size_t nLV = (size_t)NB * TP1 * ND;
  const size_t nC  = (size_t)NB * NCH * ND;
  float* lv  = (float*)d_ws;
  float* as_ = lv  + nLV;
  float* Mc  = as_ + nLV;
  float* Sc  = Mc  + nC;
  float* Mp  = Sc  + nC;
  float* Sp  = Mp  + nC;
  __bf16* Wzh = (__bf16*)(Sp + nC);
  __bf16* Wzl = Wzh + (size_t)ND * ND;
  __bf16* Whh = Wzl + (size_t)ND * ND;
  __bf16* Whl = Whh + (size_t)ND * ND;

  prep_weights<<<(ND * ND) / 256, 256, 0, stream>>>(Wz, Wzh, Wzl);
  prep_weights<<<(ND * ND) / 256, 256, 0, stream>>>(Wh, Whh, Whl);
  stage0_init<<<(NB * ND) / 256, 256, 0, stream>>>(h0, lv, as_);
  stage1_gemm<<<NB * (NT / 16), 512, 0, stream>>>(
      x, (const v16bf*)Wzh, (const v16bf*)Wzl,
      (const v16bf*)Whh, (const v16bf*)Whl, bz, bh, lv, as_);
  scan_partial<<<NB * NCH, ND, 0, stream>>>(lv, as_, Mc, Sc);
  scan_spine<<<(NB * ND) / 256, 256, 0, stream>>>(Mc, Sc, Mp, Sp);
  scan_final<<<NB * NCH, ND, 0, stream>>>(lv, as_, Mp, Sp, out);
}
